// CRF_2783138808188
// MI455X (gfx1250) — matrix operand order repaired and checked by reading, not compiled, not run on hardware
//
#include <hip/hip_runtime.h>

// ---------------------------------------------------------------------------
// CRF mean-field inference, fused exact-Gaussian filtering on gfx1250.
// C=10 classes (padded to 16), N=8192 points, D=6 features (padded to 8/32).
// Heavy math (Gram matrix + p@K accumulation) runs on v_wmma_f32_16x16x32_f16.
// Per-wave LDS transpose needs no block barrier: LDS ops are in-order per wave
// and each wave owns a private region.
// ---------------------------------------------------------------------------

typedef __attribute__((ext_vector_type(16))) _Float16 v16h;
typedef __attribute__((ext_vector_type(8)))  _Float16 v8h;
typedef __attribute__((ext_vector_type(8)))  float    v8f;

static constexpr int   NN  = 8192;
static constexpr int   CLS = 10;
static constexpr int   ITERS = 5;
static constexpr int   JSPLIT = 8;                      // j-range split for occupancy
static constexpr float L2E = 1.44269504088896340736f;   // log2(e)

// workspace layout (bytes)
static constexpr size_t OFF_F16SP = 0;                                 // 8 x N f16 (rows 3..7 zero)
static constexpr size_t OFF_F16BL = OFF_F16SP + (size_t)8 * NN * 2;    // 8 x N f16 (rows 6..7 zero)
static constexpr size_t OFF_ASCSP = OFF_F16BL + (size_t)8 * NN * 2;    // N f32: -0.5*log2e*|f|^2
static constexpr size_t OFF_ASCBL = OFF_ASCSP + (size_t)NN * 4;
static constexpr size_t OFF_PW    = OFF_ASCBL + (size_t)NN * 4;        // p, WMMA-A swizzled, 16 halves/col
static constexpr size_t OFF_OUTSP = OFF_PW    + (size_t)NN * 16 * 2;   // 16 x N f32
static constexpr size_t OFF_OUTBL = OFF_OUTSP + (size_t)16 * NN * 4;   // 16 x N f32
static constexpr size_t OFF_Q     = OFF_OUTBL + (size_t)16 * NN * 4;   // CLS x N f32 (~2 MB total)

// ---------------------------------------------------------------------------
// Prep: features -> f16 (pre-scaled by theta), plus -0.5*log2e*|f|^2 terms.
// ---------------------------------------------------------------------------
__global__ __launch_bounds__(256) void crf_prep_kernel(const float* __restrict__ feat,
                                                       char* __restrict__ ws) {
  const int j = blockIdx.x * blockDim.x + threadIdx.x;
  if (j >= NN) return;
  _Float16* fsp = (_Float16*)(ws + OFF_F16SP);
  _Float16* fbl = (_Float16*)(ws + OFF_F16BL);
  float* asp = (float*)(ws + OFF_ASCSP);
  float* abl = (float*)(ws + OFF_ASCBL);
  float f2s = 0.f, f2b = 0.f;
#pragma unroll
  for (int k = 0; k < 8; ++k) {
    const float x  = (k < 6) ? feat[(size_t)k * NN + j] : 0.0f;
    const float xs = (k < 3) ? x * 0.125f : 0.0f;   // spatial: 3 dims / theta_gamma=8
    const float xb = x;                             // bilateral: all 6 dims, theta=1
    fsp[(size_t)k * NN + j] = (_Float16)xs;
    fbl[(size_t)k * NN + j] = (_Float16)xb;
    f2s += xs * xs;  f2b += xb * xb;
  }
  asp[j] = -0.5f * L2E * f2s;
  abl[j] = -0.5f * L2E * f2b;
}

// ---------------------------------------------------------------------------
// Softmax over classes; scatter p into pre-swizzled WMMA-A (16x32 f16) layout:
// chunk = j/32; for (class m, jr=j%32): b=jr&15, hi=b>>3, lane=m+16*hi,
// vslot=((b&7)>>1)+4*(jr>>4), half=jr&1.
// ---------------------------------------------------------------------------
__global__ __launch_bounds__(256) void crf_softmax_kernel(const float* __restrict__ qsrc,
                                                          char* __restrict__ ws) {
  const int j = blockIdx.x * blockDim.x + threadIdx.x;
  if (j >= NN) return;
  _Float16* pw = (_Float16*)(ws + OFF_PW);
  float v[CLS];
  float mx = -3.4e38f;
#pragma unroll
  for (int c = 0; c < CLS; ++c) { v[c] = qsrc[(size_t)c * NN + j]; mx = fmaxf(mx, v[c]); }
  float s = 0.f;
#pragma unroll
  for (int c = 0; c < CLS; ++c) { const float e = __builtin_amdgcn_exp2f((v[c] - mx) * L2E); v[c] = e; s += e; }
  const float inv = 1.0f / s;
  const int chunk = j >> 5, jr = j & 31, b = jr & 15;
  const int hh = b >> 3;
  const int vslot = ((b & 7) >> 1) + ((jr >> 4) << 2);
  const int half = jr & 1;
#pragma unroll
  for (int m = 0; m < 16; ++m) {
    const int lane = m + (hh << 4);
    const size_t off = ((size_t)(chunk * 32 + lane) * 8 + vslot) * 2 + half;
    pw[off] = (_Float16)((m < CLS) ? v[m] * inv : 0.0f);
  }
}

// ---------------------------------------------------------------------------
// Fused Gaussian filter: out[c, i-tile] += sum_j exp(...) * p[c, j].
// Per wave: 16 output columns; loop 32 j per step:
//   2x WMMA (Gram) -> 16x v_exp_f32 -> per-wave LDS transpose -> 1x WMMA (acc).
// grid = (N/64, 2 filters, JSPLIT j-splits), block = 128 (4 waves).
// ---------------------------------------------------------------------------
__global__ __launch_bounds__(128) void crf_filter_kernel(char* __restrict__ ws) {
  const int f = blockIdx.y;
  const _Float16* feat16 = (const _Float16*)(ws + (f ? OFF_F16BL : OFF_F16SP));
  const float*    asc    = (const float*)   (ws + (f ? OFF_ASCBL : OFF_ASCSP));
  const _Float16* pw     = (const _Float16*)(ws + OFF_PW);
  float*          outb   = (float*)         (ws + (f ? OFF_OUTBL : OFF_OUTSP));

  const int lane = threadIdx.x & 31;
  const int wave = threadIdx.x >> 5;
  const int hi   = lane >> 4;
  const int m    = lane & 15;
  const int i0   = blockIdx.x * 64 + wave * 16;

  __shared__ __align__(32) _Float16 lds[4][32 * 16];   // per-wave 32(j) x 16(i) K-tile
  _Float16* myl = &lds[wave][0];

  // A operand for Gram WMMA: features of this wave's 16 i-columns (16x32 f16,
  // K slots per 16-bit A layout; only feature rows 0..7 populated).
  v16h afeat;
#pragma unroll
  for (int v = 0; v < 8; ++v) {
#pragma unroll
    for (int h = 0; h < 2; ++h) {
      const int k = (v < 4) ? (8 * hi + 2 * v + h) : (16 + 8 * hi + 2 * (v - 4) + h);
      float x = 0.0f;
      if (k < 8) x = (float)feat16[(size_t)k * NN + i0 + m];
      afeat[2 * v + h] = (_Float16)x;
    }
  }
  float ai[8];
#pragma unroll
  for (int v = 0; v < 8; ++v) ai[v] = asc[i0 + v + 8 * hi];

  v8f acc, zc;
#pragma unroll
  for (int v = 0; v < 8; ++v) { acc[v] = 0.0f; zc[v] = 0.0f; }

  // B operand registers zero-initialized ONCE; lanes >= 16 never overwrite them,
  // so no per-iteration re-zeroing is needed.
  v16h bf0, bf1;
#pragma unroll
  for (int t = 0; t < 16; ++t) { bf0[t] = (_Float16)0.0f; bf1[t] = (_Float16)0.0f; }

  const int c0 = blockIdx.z * (NN / 32 / JSPLIT);
  const int c1 = c0 + (NN / 32 / JSPLIT);
  for (int c = c0; c < c1; ++c) {
    const int j0 = c * 32;

    // B operand for Gram WMMA: lanes 0..15 hold j-column j0(+16)+lane,
    // elements 0..7 = feature rows k=0..7; all other K slots stay zero.
    if (lane < 16) {
#pragma unroll
      for (int s = 0; s < 8; ++s) {
        bf0[s] = feat16[(size_t)s * NN + j0 + lane];
        bf1[s] = feat16[(size_t)s * NN + j0 + 16 + lane];
      }
    }

    v8f g0 = __builtin_amdgcn_wmma_f32_16x16x32_f16(false, afeat, false, bf0, (short)0, zc, false, false);
    v8f g1 = __builtin_amdgcn_wmma_f32_16x16x32_f16(false, afeat, false, bf1, (short)0, zc, false, false);

    const float aj0 = asc[j0 + m];
    const float aj1 = asc[j0 + 16 + m];

    // branchless speculative prefetch of next chunk's pre-swizzled p line
    // (one line past the range lands harmlessly in the adjacent ws buffer)
    __builtin_prefetch(pw + ((size_t)(c + 1) * 32 + lane) * 16, 0, 3);

    // K[i,j] = exp2(log2e * G + a_i + a_j), to f16
    v8h t0, t1;
#pragma unroll
    for (int v = 0; v < 8; ++v) {
      t0[v] = (_Float16)__builtin_amdgcn_exp2f(fmaf(g0[v], L2E, ai[v] + aj0));
      t1[v] = (_Float16)__builtin_amdgcn_exp2f(fmaf(g1[v], L2E, ai[v] + aj1));
    }

    // transpose C/D layout -> B layout through this wave's private LDS region.
    // LDS ops are in-order within a wave; no block barrier required.
    __builtin_amdgcn_wave_barrier();
    *(v8h*)(myl + (size_t)(m     ) * 16 + 8 * hi) = t0;
    *(v8h*)(myl + (size_t)(m + 16) * 16 + 8 * hi) = t1;
    __builtin_amdgcn_wave_barrier();
    v16h bk;                                                           // column i=m, 16 j (table order)
#pragma unroll
    for (int s = 0; s < 16; ++s) {
      const int jr = (s < 8) ? (8 * hi + s) : (16 + 8 * hi + (s - 8));
      bk[s] = myl[(size_t)jr * 16 + m];
    }
    v16h ap = *(const v16h*)(pw + ((size_t)c * 32 + lane) * 16);       // pre-swizzled A
    __builtin_amdgcn_wave_barrier();
    acc = __builtin_amdgcn_wmma_f32_16x16x32_f16(false, ap, false, bk, (short)0, acc, false, false);
  }

#pragma unroll
  for (int v = 0; v < 8; ++v) {
    const int cls = v + 8 * hi;
    if (cls < CLS) unsafeAtomicAdd(&outb[(size_t)cls * NN + i0 + m], acc[v]);
  }
}

// ---------------------------------------------------------------------------
// Epilogue: mp = SW@sp + BW@bl ; q = unaries - CM@mp  (C=10, tiny per column)
// ---------------------------------------------------------------------------
__global__ __launch_bounds__(256) void crf_combine_kernel(const char* __restrict__ ws,
    const float* __restrict__ unaries, const float* __restrict__ SW,
    const float* __restrict__ BW, const float* __restrict__ CM,
    float* __restrict__ qout) {
  const int j = blockIdx.x * blockDim.x + threadIdx.x;
  if (j >= NN) return;
  const float* sp = (const float*)(ws + OFF_OUTSP);
  const float* bl = (const float*)(ws + OFF_OUTBL);
  float spv[CLS], blv[CLS], mp[CLS];
#pragma unroll
  for (int k = 0; k < CLS; ++k) { spv[k] = sp[(size_t)k * NN + j]; blv[k] = bl[(size_t)k * NN + j]; }
#pragma unroll
  for (int c = 0; c < CLS; ++c) {
    float s = 0.f;
#pragma unroll
    for (int k = 0; k < CLS; ++k) s = fmaf(SW[c * CLS + k], spv[k], fmaf(BW[c * CLS + k], blv[k], s));
    mp[c] = s;
  }
#pragma unroll
  for (int c = 0; c < CLS; ++c) {
    float s = 0.f;
#pragma unroll
    for (int k = 0; k < CLS; ++k) s = fmaf(CM[c * CLS + k], mp[k], s);
    qout[(size_t)c * NN + j] = unaries[(size_t)c * NN + j] - s;
  }
}

// ---------------------------------------------------------------------------
extern "C" void kernel_launch(void* const* d_in, const int* in_sizes, int n_in,
                              void* d_out, int out_size, void* d_ws, size_t ws_size,
                              hipStream_t stream) {
  const float* unaries = (const float*)d_in[0];
  const float* feat    = (const float*)d_in[1];
  const float* SW      = (const float*)d_in[2];
  const float* BW      = (const float*)d_in[3];
  const float* CM      = (const float*)d_in[4];
  float* out  = (float*)d_out;
  char*  ws   = (char*)d_ws;
  float* qbuf = (float*)(ws + OFF_Q);

  crf_prep_kernel<<<NN / 256, 256, 0, stream>>>(feat, ws);

  for (int it = 0; it < ITERS; ++it) {
    const float* qsrc = (it == 0) ? unaries : (const float*)qbuf;
    crf_softmax_kernel<<<NN / 256, 256, 0, stream>>>(qsrc, ws);
    hipMemsetAsync(ws + OFF_OUTSP, 0, (size_t)2 * 16 * NN * 4, stream);   // zero sp+bl outputs
    dim3 grid(NN / 64, 2, JSPLIT);
    crf_filter_kernel<<<grid, 128, 0, stream>>>(ws);
    float* qdst = (it == ITERS - 1) ? out : qbuf;
    crf_combine_kernel<<<NN / 256, 256, 0, stream>>>(ws, unaries, SW, BW, CM, qdst);
  }
}
